// RNNDecoder_26027501814589
// MI455X (gfx1250) — matrix operand order repaired and checked by reading, not compiled, not run on hardware
//
#include <hip/hip_runtime.h>

typedef __attribute__((ext_vector_type(16))) __bf16 v16bf;
typedef __attribute__((ext_vector_type(8)))  __bf16 v8bf;
typedef __attribute__((ext_vector_type(8)))  float  v8f;

#define DEV __device__ __forceinline__

namespace {
constexpr int Bn = 2, Nn = 100, Tn = 15, Fdim = 6, Kn = 4, Hn = 128;
constexpr int En = Nn * (Nn - 1);        // 9900 edges
constexpr int ET = (En + 15) / 16;       // 619 edge tiles of 16
constexpr int NTL = (Nn + 15) / 16;      // 7 node tiles of 16
constexpr int Sn = Tn - 1;               // 14 time steps
}

DEV __bf16 f2bf(float f) {
  union { float f; unsigned u; } cv; cv.f = f;
  unsigned u = cv.u;
  u += 0x7FFFu + ((u >> 16) & 1u);           // round-to-nearest-even
  union { unsigned short s; __bf16 b; } ov;
  ov.s = (unsigned short)(u >> 16);
  return ov.b;
}

// branch-free transcendentals: v_exp_f32 + v_rcp_f32, no exec-divergent slow path
DEV float tanh_fast(float x) {
  float ax = fabsf(x);
  float t  = __expf(-2.0f * ax);
  float r  = (1.0f - t) * __builtin_amdgcn_rcpf(1.0f + t);
  return copysignf(r, x);
}
DEV float sigmoid_fast(float x) {
  return __builtin_amdgcn_rcpf(1.0f + __expf(-x));
}

DEV v8f wmma_bf16(v16bf a, v16bf b, v8f c) {
  return __builtin_amdgcn_wmma_f32_16x16x32_bf16(
      false, a, false, b, (short)0, c, false, false);
}

// A-fragment (16x32 bf16, M=row, K=col) from an LDS tile with leading dim `ld`
// ISA 7.12.2: lane m = l&15, half = l>>4;
// elems 0..7 -> K = kc*32 + half*8 + j ; elems 8..15 -> K = kc*32 + 16 + half*8 + j
DEV v16bf load_a_frag(const __bf16* base, int ld, int kc, int lane) {
  int m = lane & 15, half = lane >> 4;
  const __bf16* p = base + m * ld + kc * 32 + half * 8;
  v8bf lo = *(const v8bf*)(p);
  v8bf hi = *(const v8bf*)(p + 16);
  v16bf a;
#pragma unroll
  for (int j = 0; j < 8; ++j) { a[j] = lo[j]; a[8 + j] = hi[j]; }
  return a;
}

// ---------------------------------------------------------------------------
// Prep kernels
// ---------------------------------------------------------------------------

__global__ void build_idx_kernel(const float* __restrict__ rel_rec,
                                 const float* __restrict__ rel_send,
                                 int* __restrict__ recv_idx,
                                 int* __restrict__ send_idx) {
  int e = blockIdx.x * blockDim.x + threadIdx.x;
  if (e >= En) return;
  int r = 0, s = 0;
  for (int n = 0; n < Nn; ++n) {
    if (rel_rec [e * Nn + n] > 0.5f) r = n;
    if (rel_send[e * Nn + n] > 0.5f) s = n;
  }
  recv_idx[e] = r;
  send_idx[e] = s;
}

// Pack a [Kdim x 128] row-major f32 weight into WMMA B-fragment order:
// dst[((nt*KC + kc)*32 + lane)*16 + j] = src[(kc*32 + (j>>3)*16 + (lane>>4)*8 + (j&7))*128 + nt*16 + (lane&15)]
__global__ void pack_w_kernel(const float* __restrict__ src,
                              __bf16* __restrict__ dst, int Kdim) {
  int i = blockIdx.x * blockDim.x + threadIdx.x;
  int total = Kdim * Hn;
  if (i >= total) return;
  int KC   = Kdim >> 5;
  int j    = i & 15;
  int lane = (i >> 4) & 31;
  int rest = i >> 9;
  int kc = rest % KC;
  int nt = rest / KC;
  int row = kc * 32 + (j >> 3) * 16 + (lane >> 4) * 8 + (j & 7);
  int col = nt * 16 + (lane & 15);
  dst[i] = f2bf(src[row * Hn + col]);
}

__global__ void init_state_kernel(const float* __restrict__ data,
                                  float* __restrict__ hidden,
                                  float* __restrict__ prev_pred) {
  int i = blockIdx.x * blockDim.x + threadIdx.x;
  if (i < Bn * Nn * Hn) hidden[i] = 0.0f;
  if (i < Bn * Nn * Fdim) {
    int b = i / (Nn * Fdim);
    int rem = i % (Nn * Fdim);
    int n = rem / Fdim, f = rem % Fdim;
    prev_pred[i] = data[((b * Nn + n) * Tn + 0) * Fdim + f];
  }
}

__global__ void zero_agg_kernel(float* __restrict__ agg) {
  int i = blockIdx.x * blockDim.x + threadIdx.x;
  if (i < Bn * Nn * Hn) agg[i] = 0.0f;
}

// ---------------------------------------------------------------------------
// Edge-message kernel: one block = one 16-edge tile (both MLP layers fused).
// 4 waves; wave w handles edge type k=w. WMMA bf16, f32 accumulate.
// ---------------------------------------------------------------------------
__global__ __launch_bounds__(128) void edge_kernel(
    const float* __restrict__ hidden, const float* __restrict__ rel_type,
    const int* __restrict__ recv_idx, const int* __restrict__ send_idx,
    const __bf16* __restrict__ W1p, const __bf16* __restrict__ W2p,
    const float* __restrict__ b1, const float* __restrict__ b2,
    float* __restrict__ agg) {
  __shared__ __bf16 sPre[16 * 256];       // pre_msg tile (send|recv), bf16 (8KB)
  __shared__ __bf16 sM1[Kn][16 * 128];    // layer-1 activations per type (16KB)
  __shared__ float  sM2[Kn][16 * 128];    // scaled layer-2 output per type (32KB)

  int b    = blockIdx.x / ET;
  int tile = blockIdx.x % ET;
  int e0   = tile * 16;
  int tid  = threadIdx.x;
  int lane = tid & 31;
  int k    = tid >> 5;                    // wave index == edge type
  int half = lane >> 4, ncol = lane & 15;

  // gather pre_msg = [hidden[send] | hidden[recv]]   (tid = h index, H=128)
#pragma unroll 4
  for (int m = 0; m < 16; ++m) {
    int e = e0 + m;
    float vs = 0.0f, vr = 0.0f;
    if (e < En) {
      vs = hidden[(b * Nn + send_idx[e]) * Hn + tid];
      vr = hidden[(b * Nn + recv_idx[e]) * Hn + tid];
    }
    sPre[m * 256 + tid]       = f2bf(vs);
    sPre[m * 256 + 128 + tid] = f2bf(vr);
  }
  __syncthreads();

  // ---- layer 1: m1 = tanh(pre_msg @ W1[k] + b1[k])  ([16x256]@[256x128])
  for (int nt = 0; nt < 8; ++nt) {
    v8f acc = {};
#pragma unroll
    for (int kc = 0; kc < 8; ++kc) {
      v16bf a  = load_a_frag(sPre, 256, kc, lane);
      v16bf bm = *(const v16bf*)&W1p[(((k * 8 + nt) * 8 + kc) * 32 + lane) * 16];
      acc = wmma_bf16(a, bm, acc);
    }
    float bias = b1[k * Hn + nt * 16 + ncol];
#pragma unroll
    for (int v = 0; v < 8; ++v) {
      int m = v + 8 * half;
      sM1[k][m * 128 + nt * 16 + ncol] = f2bf(tanh_fast(acc[v] + bias));
    }
  }
  // same-wave LDS RAW: DS ops are in-order per wave; compiler inserts dscnt waits.

  // ---- layer 2: m2 = tanh(m1 @ W2[k] + b2[k]); scale by rel_type/(K*F)
  const float inv_norm = 1.0f / (float)(Kn * Fdim);
  for (int nt = 0; nt < 8; ++nt) {
    v8f acc = {};
#pragma unroll
    for (int kc = 0; kc < 4; ++kc) {
      v16bf a  = load_a_frag(sM1[k], 128, kc, lane);
      v16bf bm = *(const v16bf*)&W2p[(((k * 8 + nt) * 4 + kc) * 32 + lane) * 16];
      acc = wmma_bf16(a, bm, acc);
    }
    float bias = b2[k * Hn + nt * 16 + ncol];
#pragma unroll
    for (int v = 0; v < 8; ++v) {
      int m = v + 8 * half;
      int e = e0 + m;
      float rt = (e < En) ? rel_type[(b * En + e) * Kn + k] : 0.0f;
      sM2[k][m * 128 + nt * 16 + ncol] =
          tanh_fast(acc[v] + bias) * rt * inv_norm;   // plain ds_store_b32
    }
  }
  __syncthreads();

  // reduce over edge types in registers, one global atomic per element
  for (int i = tid; i < 16 * 128; i += 128) {
    int m = i >> 7, h = i & 127;
    int e = e0 + m;
    if (e < En) {
      float s = sM2[0][i] + sM2[1][i] + sM2[2][i] + sM2[3][i];
      atomicAdd(&agg[(b * Nn + recv_idx[e]) * Hn + h], s);
    }
  }
}

// ---------------------------------------------------------------------------
// Node kernel: GRU update + output MLP for a 16-node tile. 4 waves.
// ---------------------------------------------------------------------------

// [16x128] bf16 (LDS) @ packed [128x128] -> f32 LDS. Waves split the 8 n-tiles.
DEV void gemm16x128(const __bf16* sA, const __bf16* __restrict__ Wp,
                    float* sOut, int wave, int lane) {
  int half = lane >> 4, ncol = lane & 15;
#pragma unroll
  for (int t = 0; t < 2; ++t) {
    int nt = wave * 2 + t;
    v8f acc = {};
#pragma unroll
    for (int kc = 0; kc < 4; ++kc) {
      v16bf a  = load_a_frag(sA, 128, kc, lane);
      v16bf bm = *(const v16bf*)&Wp[((nt * 4 + kc) * 32 + lane) * 16];
      acc = wmma_bf16(a, bm, acc);
    }
#pragma unroll
    for (int v = 0; v < 8; ++v)
      sOut[(v + 8 * half) * 128 + nt * 16 + ncol] = acc[v];
  }
}

__global__ __launch_bounds__(128) void node_kernel(
    const float* __restrict__ data, const int* __restrict__ pred_steps_p, int step,
    const float* __restrict__ agg,
    const __bf16* __restrict__ Whrp, const __bf16* __restrict__ Whip,
    const __bf16* __restrict__ Whhp, const __bf16* __restrict__ Wo1p,
    const __bf16* __restrict__ Wo2p,
    const float* __restrict__ Wir, const float* __restrict__ bir,
    const float* __restrict__ Wii, const float* __restrict__ bii,
    const float* __restrict__ Win, const float* __restrict__ binp,
    const float* __restrict__ bo1, const float* __restrict__ bo2,
    const float* __restrict__ Wo3, const float* __restrict__ bo3,
    float* __restrict__ hidden, float* __restrict__ prev_pred,
    float* __restrict__ out) {
  __shared__ __bf16 sA[16 * 128];       // bf16 GEMM input (agg, then relu(p1))
  __shared__ __bf16 sHn[16 * 128];      // bf16 hidden_new
  __shared__ float  sR[16 * 128], sI[16 * 128], sNN[16 * 128], sP[16 * 128];
  __shared__ float  sIns[16 * Fdim];

  int b    = blockIdx.x / NTL;
  int tile = blockIdx.x % NTL;
  int n0   = tile * 16;
  int tid  = threadIdx.x;
  int lane = tid & 31, wave = tid >> 5;

  int ps = pred_steps_p[0]; if (ps < 1) ps = 1;
  bool use_gt = (step % ps) == 0;

  if (tid < 16 * Fdim) {
    int m = tid / Fdim, f = tid % Fdim, n = n0 + m;
    float v = 0.0f;
    if (n < Nn)
      v = use_gt ? data[((b * Nn + n) * Tn + step) * Fdim + f]
                 : prev_pred[(b * Nn + n) * Fdim + f];
    sIns[tid] = v;
  }
  for (int m = 0; m < 16; ++m) {
    int n = n0 + m;
    float v = (n < Nn) ? agg[(b * Nn + n) * Hn + tid] : 0.0f;
    sA[m * 128 + tid] = f2bf(v);
  }
  __syncthreads();

  gemm16x128(sA, Whrp, sR, wave, lane);
  gemm16x128(sA, Whip, sI, wave, lane);
  gemm16x128(sA, Whhp, sNN, wave, lane);
  __syncthreads();

  // element-wise GRU gates (tiny F=6 matvecs done scalar)
  for (int i = tid; i < 16 * 128; i += 128) {
    int m = i >> 7, h = i & 127, n = n0 + m;
    float xr = bir[h], xi = bii[h], xn = binp[h];
#pragma unroll
    for (int f = 0; f < Fdim; ++f) {
      float iv = sIns[m * Fdim + f];
      xr += iv * Wir[f * Hn + h];
      xi += iv * Wii[f * Hn + h];
      xn += iv * Win[f * Hn + h];
    }
    float r  = sigmoid_fast(xr + sR[i]);
    float ig = sigmoid_fast(xi + sI[i]);
    float nn = tanh_fast(xn + r * sNN[i]);
    float hold = (n < Nn) ? hidden[(b * Nn + n) * Hn + h] : 0.0f;
    float hnew = (1.0f - ig) * nn + ig * hold;
    if (n < Nn) hidden[(b * Nn + n) * Hn + h] = hnew;
    sHn[i] = f2bf(hnew);
  }
  __syncthreads();

  gemm16x128(sHn, Wo1p, sP, wave, lane);
  __syncthreads();
  for (int i = tid; i < 16 * 128; i += 128) {
    float v = sP[i] + bo1[i & 127];
    sA[i] = f2bf(v > 0.0f ? v : 0.0f);
  }
  __syncthreads();
  gemm16x128(sA, Wo2p, sP, wave, lane);
  __syncthreads();
  for (int i = tid; i < 16 * 128; i += 128) {
    float v = sP[i] + bo2[i & 127];
    sP[i] = v > 0.0f ? v : 0.0f;
  }
  __syncthreads();

  // pred = ins + p @ Wo3 + bo3   ([16x128]@[128x6] — scalar dot per output)
  if (tid < 16 * Fdim) {
    int m = tid / Fdim, f = tid % Fdim, n = n0 + m;
    if (n < Nn) {
      float acc = bo3[f];
      for (int h = 0; h < Hn; ++h) acc += sP[m * 128 + h] * Wo3[h * Fdim + f];
      float pred = sIns[tid] + acc;
      out[((b * Nn + n) * Sn + step) * Fdim + f] = pred;
      prev_pred[(b * Nn + n) * Fdim + f] = pred;
    }
  }
}

// ---------------------------------------------------------------------------
// Host launcher
// ---------------------------------------------------------------------------
static inline size_t alignUp(size_t x, size_t a) { return (x + a - 1) & ~(a - 1); }

extern "C" void kernel_launch(void* const* d_in, const int* in_sizes, int n_in,
                              void* d_out, int out_size, void* d_ws, size_t ws_size,
                              hipStream_t stream) {
  (void)in_sizes; (void)n_in; (void)out_size; (void)ws_size;
  const float* data     = (const float*)d_in[0];
  const float* rel_type = (const float*)d_in[1];
  const float* rel_rec  = (const float*)d_in[2];
  const float* rel_send = (const float*)d_in[3];
  const int*   pred_steps = (const int*)d_in[4];
  const float* W1 = (const float*)d_in[5];
  const float* b1 = (const float*)d_in[6];
  const float* W2 = (const float*)d_in[7];
  const float* b2 = (const float*)d_in[8];
  const float* Whr = (const float*)d_in[9];
  const float* Whi = (const float*)d_in[10];
  const float* Whh = (const float*)d_in[11];
  const float* Wir = (const float*)d_in[12];
  const float* bir = (const float*)d_in[13];
  const float* Wii = (const float*)d_in[14];
  const float* bii = (const float*)d_in[15];
  const float* Win = (const float*)d_in[16];
  const float* binp = (const float*)d_in[17];
  const float* Wo1 = (const float*)d_in[18];
  const float* bo1 = (const float*)d_in[19];
  const float* Wo2 = (const float*)d_in[20];
  const float* bo2 = (const float*)d_in[21];
  const float* Wo3 = (const float*)d_in[22];
  const float* bo3 = (const float*)d_in[23];
  float* out = (float*)d_out;

  char* base = (char*)d_ws;
  size_t off = 0;
  auto carve = [&](size_t bytes) -> void* {
    void* r = base + off;
    off = alignUp(off + bytes, 256);
    return r;
  };
  float* hidden    = (float*)carve((size_t)Bn * Nn * Hn * 4);
  float* agg       = (float*)carve((size_t)Bn * Nn * Hn * 4);
  float* prev_pred = (float*)carve((size_t)Bn * Nn * Fdim * 4);
  int*   recv_idx  = (int*)carve((size_t)En * 4);
  int*   send_idx  = (int*)carve((size_t)En * 4);
  __bf16* W1p  = (__bf16*)carve((size_t)Kn * 2 * Hn * Hn * 2);
  __bf16* W2p  = (__bf16*)carve((size_t)Kn * Hn * Hn * 2);
  __bf16* Whrp = (__bf16*)carve((size_t)Hn * Hn * 2);
  __bf16* Whip = (__bf16*)carve((size_t)Hn * Hn * 2);
  __bf16* Whhp = (__bf16*)carve((size_t)Hn * Hn * 2);
  __bf16* Wo1p = (__bf16*)carve((size_t)Hn * Hn * 2);
  __bf16* Wo2p = (__bf16*)carve((size_t)Hn * Hn * 2);

  // ---- one-time prep (deterministic; re-run every launch) ----
  build_idx_kernel<<<(En + 127) / 128, 128, 0, stream>>>(rel_rec, rel_send,
                                                         recv_idx, send_idx);
  for (int k = 0; k < Kn; ++k) {
    pack_w_kernel<<<(2 * Hn * Hn + 127) / 128, 128, 0, stream>>>(
        W1 + (size_t)k * 2 * Hn * Hn, W1p + (size_t)k * 2 * Hn * Hn, 2 * Hn);
    pack_w_kernel<<<(Hn * Hn + 127) / 128, 128, 0, stream>>>(
        W2 + (size_t)k * Hn * Hn, W2p + (size_t)k * Hn * Hn, Hn);
  }
  pack_w_kernel<<<(Hn * Hn + 127) / 128, 128, 0, stream>>>(Whr, Whrp, Hn);
  pack_w_kernel<<<(Hn * Hn + 127) / 128, 128, 0, stream>>>(Whi, Whip, Hn);
  pack_w_kernel<<<(Hn * Hn + 127) / 128, 128, 0, stream>>>(Whh, Whhp, Hn);
  pack_w_kernel<<<(Hn * Hn + 127) / 128, 128, 0, stream>>>(Wo1, Wo1p, Hn);
  pack_w_kernel<<<(Hn * Hn + 127) / 128, 128, 0, stream>>>(Wo2, Wo2p, Hn);
  init_state_kernel<<<(Bn * Nn * Hn + 127) / 128, 128, 0, stream>>>(
      data, hidden, prev_pred);

  // ---- time-step loop: zero agg -> edge messages (WMMA) -> node update (WMMA)
  for (int step = 0; step < Sn; ++step) {
    zero_agg_kernel<<<(Bn * Nn * Hn + 127) / 128, 128, 0, stream>>>(agg);
    edge_kernel<<<Bn * ET, 128, 0, stream>>>(hidden, rel_type, recv_idx,
                                             send_idx, W1p, W2p, b1, b2, agg);
    node_kernel<<<Bn * NTL, 128, 0, stream>>>(
        data, pred_steps, step, agg, Whrp, Whip, Whhp, Wo1p, Wo2p,
        Wir, bir, Wii, bii, Win, binp, bo1, bo2, Wo3, bo3,
        hidden, prev_pred, out);
  }
}
